// BatchInvariantAttention_83021717831914
// MI455X (gfx1250) — matrix operand order repaired and checked by reading, not compiled, not run on hardware
//
#include <hip/hip_runtime.h>

typedef __attribute__((ext_vector_type(16))) _Float16 v16h;
typedef __attribute__((ext_vector_type(8)))  _Float16 v8h;
typedef __attribute__((ext_vector_type(4)))  _Float16 v4h;
typedef __attribute__((ext_vector_type(2)))  _Float16 v2h;
typedef __attribute__((ext_vector_type(8)))  float    v8f;
typedef __attribute__((ext_vector_type(4)))  float    f4;

#define S_LEN 2048
#define DH    64
#define QT    64    // queries per block (4 waves x 16)
#define KT    32    // keys per iteration
#define KSTR  40    // padded f16 row stride for K tile (80B: 16B-aligned, conflict-free)
#define VSTR  40    // padded f16 row stride for transposed V tile

union H16 { v16h v; v8h h[2]; v2h p[8]; };
union H4  { v4h  v; v2h p[2]; };

static __device__ __forceinline__ v2h pk2(float a, float b) {
    return __builtin_bit_cast(v2h, __builtin_amdgcn_cvt_pkrtz(a, b));
}

__global__ __launch_bounds__(128)
void fattn_wmma_kernel(const float* __restrict__ Q, const float* __restrict__ K,
                       const float* __restrict__ V, float* __restrict__ O)
{
    __shared__ __align__(16) _Float16 Qs[QT * DH];       // [q][d]  8KB (used once)
    __shared__ __align__(16) _Float16 Ks[KT * KSTR];     // [k][d]  2.5KB padded
    __shared__ __align__(16) _Float16 Vts[DH * VSTR];    // [d][k]  5KB padded

    const int tid  = threadIdx.x;
    const int wave = tid >> 5;
    const int lane = tid & 31;
    const int col  = lane & 15;
    const int hi16 = lane >> 4;

    const int bh    = blockIdx.x >> 5;        // S/QT = 32 query blocks per head
    const int qblk  = blockIdx.x & 31;
    const long hoff = (long)bh * S_LEN * DH;
    const int qbase = qblk * QT;

    const float Cc = 0.18033688011112042f;    // (1/sqrt(D)) * log2(e), folded into Q

    // ---------------- stage Q tile (pre-scaled, fp32 -> fp16) ----------------
    {
        const f4* qg = (const f4*)(Q + hoff + (long)qbase * DH);
        #pragma unroll
        for (int i = 0; i < 8; ++i) {                 // 1024 float4 / 128 threads
            int idx = tid + i * 128;
            f4 x = qg[idx];
            int row = idx >> 4, c4 = (idx & 15) << 2;
            H4 u;
            u.p[0] = pk2(x.x * Cc, x.y * Cc);
            u.p[1] = pk2(x.z * Cc, x.w * Cc);
            *(v4h*)&Qs[row * DH + c4] = u.v;
        }
    }
    __syncthreads();

    // Q^T B-operands, kept in registers for the whole kernel.
    H16 qb[2];
    {
        const int qrow = wave * 16 + col;
        #pragma unroll
        for (int c = 0; c < 2; ++c) {
            int off = qrow * DH + c * 32 + hi16 * 8;
            qb[c].h[0] = *(const v8h*)&Qs[off];
            qb[c].h[1] = *(const v8h*)&Qs[off + 16];
        }
    }

    // Constant all-ones A operand: row-sum tile => l recurrence done by the matrix pipe.
    H16 ones;
    #pragma unroll
    for (int j = 0; j < 8; ++j) ones.p[j] = pk2(1.0f, 1.0f);

    // Remapped key rows (matrix row m -> memory key), chosen so every lane's
    // score C-registers are exactly its P^T B-operand slot (no cross-lane traffic):
    //   tile0: m 0..7 -> keys 0..7,  m 8..15 -> keys 24..31
    //   tile1: m 0..7 -> keys 8..15, m 8..15 -> keys 16..23
    const int r0 = col + ((col & 8) << 1);    // {0..7, 24..31}
    const int r1 = col + 8;                   // {8..23}
    const _Float16* kap0 = &Ks[r0 * KSTR + hi16 * 8];
    const _Float16* kap1 = &Ks[r1 * KSTR + hi16 * 8];
    // V^T A-operand key offsets absorbing the matching permutation:
    const int voff0 = hi16 * 24;              // keys {0..7}  / {24..31}
    const int voff1 = 8 + hi16 * 8;           // keys {8..15} / {16..23}

    int lrow[4], lc4[4];
    #pragma unroll
    for (int i = 0; i < 4; ++i) { int idx = tid + i * 128; lrow[i] = idx >> 4; lc4[i] = (idx & 15) << 2; }

    // ---------------- prefetch first K/V tile into registers ----------------
    f4 pkr[4], pvr[4];
    {
        const f4* kg = (const f4*)(K + hoff);
        const f4* vg = (const f4*)(V + hoff);
        #pragma unroll
        for (int i = 0; i < 4; ++i) { int idx = tid + i * 128; pkr[i] = kg[idx]; pvr[i] = vg[idx]; }
    }

    // acc[0..3]: O^T tiles (d blocks of 16) x q(16); acc[4]: running softmax denominator
    v8f acc[5] = {};
    float m_run = -1e30f;

    for (int kb = 0; kb < S_LEN; kb += KT) {
        __syncthreads();                      // previous tile fully consumed

        // spill prefetched tile to LDS (fp32 -> fp16, V transposed)
        #pragma unroll
        for (int i = 0; i < 4; ++i) {
            H4 u;
            u.p[0] = pk2(pkr[i].x, pkr[i].y);
            u.p[1] = pk2(pkr[i].z, pkr[i].w);
            *(v4h*)&Ks[lrow[i] * KSTR + lc4[i]] = u.v;
            Vts[(lc4[i] + 0) * VSTR + lrow[i]] = (_Float16)pvr[i].x;
            Vts[(lc4[i] + 1) * VSTR + lrow[i]] = (_Float16)pvr[i].y;
            Vts[(lc4[i] + 2) * VSTR + lrow[i]] = (_Float16)pvr[i].z;
            Vts[(lc4[i] + 3) * VSTR + lrow[i]] = (_Float16)pvr[i].w;
        }
        __syncthreads();

        // issue next tile's global loads early (latency hidden under compute)
        if (kb + KT < S_LEN) {
            const f4* kg = (const f4*)(K + hoff + (long)(kb + KT) * DH);
            const f4* vg = (const f4*)(V + hoff + (long)(kb + KT) * DH);
            #pragma unroll
            for (int i = 0; i < 4; ++i) { int idx = tid + i * 128; pkr[i] = kg[idx]; pvr[i] = vg[idx]; }
        }

        // ------------- S^T = K * Q^T : two 16(k) x 16(q) tiles, 4 WMMA -------------
        v8f st0 = {}, st1 = {};
        #pragma unroll
        for (int c = 0; c < 2; ++c) {
            H16 ka;
            ka.h[0] = *(const v8h*)&kap0[c * 32];
            ka.h[1] = *(const v8h*)&kap0[c * 32 + 16];
            st0 = __builtin_amdgcn_wmma_f32_16x16x32_f16(false, ka.v, false, qb[c].v,
                                                         (short)0, st0, false, false);
        }
        #pragma unroll
        for (int c = 0; c < 2; ++c) {
            H16 ka;
            ka.h[0] = *(const v8h*)&kap1[c * 32];
            ka.h[1] = *(const v8h*)&kap1[c * 32 + 16];
            st1 = __builtin_amdgcn_wmma_f32_16x16x32_f16(false, ka.v, false, qb[c].v,
                                                         (short)0, st1, false, false);
        }

        // ------------- online softmax (scores already in log2 domain) -------------
        float mx = fmaxf(st0[0], st1[0]);
        #pragma unroll
        for (int j = 1; j < 8; ++j) mx = fmaxf(mx, fmaxf(st0[j], st1[j]));
        mx = fmaxf(mx, __shfl_xor(mx, 16, 32));

        // Rescale only when some query's running max actually changed.
        // Condition is wave-uniform (ballot) -> s_cbranch, EXEC untouched (WMMA-safe).
        if (__ballot(mx > m_run) != 0ull) {
            const float m_new = fmaxf(m_run, mx);
            const float alpha = __builtin_amdgcn_exp2f(m_run - m_new);
            #pragma unroll
            for (int t = 0; t < 5; ++t)
                #pragma unroll
                for (int j = 0; j < 8; ++j) acc[t][j] *= alpha;
            m_run = m_new;
        }

        float p0[8], p1[8];
        #pragma unroll
        for (int j = 0; j < 8; ++j) p0[j] = __builtin_amdgcn_exp2f(st0[j] - m_run);
        #pragma unroll
        for (int j = 0; j < 8; ++j) p1[j] = __builtin_amdgcn_exp2f(st1[j] - m_run);

        // P^T B-operand: purely local, no shuffles/selects (layout by construction)
        H16 pb;
        #pragma unroll
        for (int j = 0; j < 4; ++j) {
            pb.p[j]     = pk2(p0[2*j], p0[2*j+1]);
            pb.p[4 + j] = pk2(p1[2*j], p1[2*j+1]);
        }

        // ------------- O^T += V^T * P^T : 4 WMMA, plus l-tile (ones * P^T) -------------
        #pragma unroll
        for (int t = 0; t < 4; ++t) {
            const _Float16* vap = &Vts[(t * 16 + col) * VSTR];
            H16 va;
            va.h[0] = *(const v8h*)&vap[voff0];
            va.h[1] = *(const v8h*)&vap[voff1];
            acc[t] = __builtin_amdgcn_wmma_f32_16x16x32_f16(false, va.v, false, pb.v,
                                                            (short)0, acc[t], false, false);
        }
        acc[4] = __builtin_amdgcn_wmma_f32_16x16x32_f16(false, ones.v, false, pb.v,
                                                        (short)0, acc[4], false, false);
    }

    // ------------- normalize and store (O^T C-layout -> 2x16B contiguous per lane) -------------
    const float inv_l = __builtin_amdgcn_rcpf(acc[4][0]);   // full 32-key sums, all rows equal
    const int qrow = qbase + wave * 16 + col;
    float* outp = O + hoff + (long)qrow * DH;
    #pragma unroll
    for (int t = 0; t < 4; ++t) {
        const int d0 = t * 16 + hi16 * 8;
        f4 r0v, r1v;
        r0v.x = acc[t][0] * inv_l; r0v.y = acc[t][1] * inv_l;
        r0v.z = acc[t][2] * inv_l; r0v.w = acc[t][3] * inv_l;
        r1v.x = acc[t][4] * inv_l; r1v.y = acc[t][5] * inv_l;
        r1v.z = acc[t][6] * inv_l; r1v.w = acc[t][7] * inv_l;
        *(f4*)(outp + d0)     = r0v;
        *(f4*)(outp + d0 + 4) = r1v;
    }
}

extern "C" void kernel_launch(void* const* d_in, const int* in_sizes, int n_in,
                              void* d_out, int out_size, void* d_ws, size_t ws_size,
                              hipStream_t stream) {
    const float* Q = (const float*)d_in[0];
    const float* K = (const float*)d_in[1];
    const float* V = (const float*)d_in[2];
    float* O = (float*)d_out;
    dim3 grid(4 * 16 * (S_LEN / QT));   // 2048 workgroups
    dim3 block(128);                    // 4 waves
    hipLaunchKernelGGL(fattn_wmma_kernel, grid, block, 0, stream, Q, K, V, O);
}
